// SSNetCell_50663434223896
// MI455X (gfx1250) — matrix operand order repaired and checked by reading, not compiled, not run on hardware
//
#include <hip/hip_runtime.h>
#include <hip/hip_bf16.h>
#include <stdint.h>

// ---------------------------------------------------------------------------
// Conditional 2-layer LSTM (B=512, T=64, IN=64, H=1024, OUT=128) for gfx1250.
// bf16 WMMA GEMMs; weights staged Global->LDS with async-to-LDS (ASYNCcnt)
// double buffering, shared by 4 waves per block.
// ---------------------------------------------------------------------------

#ifndef USE_ASYNC_LDS
#define USE_ASYNC_LDS 1   // 0 => plain global_load + ds_store fallback
#endif

typedef __attribute__((ext_vector_type(16))) __bf16 v16bf;
typedef __attribute__((ext_vector_type(8)))  float  v8f;

#define HD   1024
#define BSZ  512
#define TSEQ 64
#define INF_ 64

// ---- fragment loaders (contiguous k = 16*(lane>>4) + j, wave32) ---------------
__device__ __forceinline__ v16bf load_frag_pair(const __bf16* p0, const __bf16* p1) {
  union { uint4 u[2]; v16bf v; } t;
  t.u[0] = *reinterpret_cast<const uint4*>(p0);
  t.u[1] = *reinterpret_cast<const uint4*>(p1);
  return t.v;
}
__device__ __forceinline__ v16bf load_a_frag(const __bf16* row, int kh) {
  // A 16x32: lane l holds row m=l&15; elems 0..15 = K[kh*16..+16)
  return load_frag_pair(row + kh * 16, row + kh * 16 + 8);
}
__device__ __forceinline__ float sigmoidf_(float x) {
  return 1.0f / (1.0f + __expf(-x));
}

// ---- prep kernels ----------------------------------------------------------
__global__ void cvt_bf16_kernel(const float* __restrict__ src,
                                __bf16* __restrict__ dst, long n) {
  long i = (long)blockIdx.x * blockDim.x + threadIdx.x;
  if (i < n) dst[i] = (__bf16)src[i];
}

__global__ void zero_u32_kernel(uint32_t* __restrict__ p, long n) {
  long i = (long)blockIdx.x * blockDim.x + threadIdx.x;
  if (i < n) p[i] = 0u;
}

// Pack [Wx | Wh] (K-concatenated, transposed) into per-tile B-fragment order:
// element (tile, lane, j) = Wcat[k][n], k = kt*32 + (lane>>4)*16 + j,
// n = ntile*16 + (lane&15), tile = kt*NT + ntile, 512 bf16 (1KB) per tile.
__global__ void pack_weights_kernel(const float* __restrict__ srcA, int lda, int offA, int K0,
                                    const float* __restrict__ srcB, int ldb,
                                    __bf16* __restrict__ dst, int NT, long total) {
  long idx = (long)blockIdx.x * blockDim.x + threadIdx.x;
  if (idx >= total) return;
  long tile  = idx >> 9;
  int within = (int)(idx & 511);
  int l = within >> 4, j = within & 15;
  int kt = (int)(tile / NT), ntile = (int)(tile % NT);
  int k = (kt << 5) + ((l >> 4) << 4) + j;
  int n = (ntile << 4) + (l & 15);
  float v = (k < K0) ? srcA[(long)n * lda + offA + k]
                     : srcB[(long)n * ldb + (k - K0)];
  dst[idx] = (__bf16)v;
}

// ---- one (timestep, layer): gates GEMM + LSTM epilogue --------------------
// Block = 4 waves (128 thr), M=128 (32 rows per wave), N = 16 hidden units
// across all 4 gates. The 4 gate B-tiles per K-step (4KB) are staged into
// LDS once per block (async global->LDS, double buffered) and shared.
__global__ __launch_bounds__(128)
void lstm_gate_step(const __bf16* __restrict__ A0, long a0s, int K0,
                    const __bf16* __restrict__ A1, long a1s, int K1,
                    const __bf16* __restrict__ Wpk,
                    const float* __restrict__ bias,             // 4*HD
                    const float* __restrict__ wxc0, long wxc0s, // rank-1 flag col (or null)
                    const float* __restrict__ flags, long flagss,
                    const float* __restrict__ c_in,
                    const float* __restrict__ h_in,
                    float* __restrict__ c_out,
                    float* __restrict__ h_out,
                    __bf16* __restrict__ hbf_out) {
  __shared__ __bf16 Bsh[2][4][512];   // [buf][gate][tile], 8KB

  const int nh   = blockIdx.x;        // 0..63 hidden-unit tile
  const int mb   = blockIdx.y;        // 0..3  (128 batch rows per block)
  const int tid  = threadIdx.x;
  const int wv   = tid >> 5;          // wave id: also the gate tile it stages
  const int lane = tid & 31;
  const int mrow = lane & 15, kh = lane >> 4;
  const int KT   = (K0 + K1) >> 5;

  // staging source: this thread copies 32B of gate tile g=wv, chunk=lane
  const long tileStride = (long)256 << 9;  // elements per K-step (256 tiles * 512)
  const __bf16* gsrc0 = Wpk + (((long)(wv << 6) + nh) << 9) + (lane << 4);
  uint32_t ldsoff[2];
  ldsoff[0] = (uint32_t)(uintptr_t)&Bsh[0][wv][lane << 4];
  ldsoff[1] = (uint32_t)(uintptr_t)&Bsh[1][wv][lane << 4];

  v8f acc[4][2];
#pragma unroll
  for (int g = 0; g < 4; ++g)
#pragma unroll
    for (int mi = 0; mi < 2; ++mi) acc[g][mi] = (v8f)0.0f;

  // ---- stage one K-step's 4 gate tiles into LDS buffer ----
  auto stage = [&](int kt, int buf) {
    const __bf16* g = gsrc0 + (long)kt * tileStride;
#if USE_ASYNC_LDS
    asm volatile("global_load_async_to_lds_b128 %0, %1, off"
                 :: "v"(ldsoff[buf]), "v"(g) : "memory");
    asm volatile("global_load_async_to_lds_b128 %0, %1, off offset:16"
                 :: "v"(ldsoff[buf]), "v"(g) : "memory");
#else
    uint4 d0 = *reinterpret_cast<const uint4*>(g);
    uint4 d1 = *reinterpret_cast<const uint4*>(g + 8);
    *reinterpret_cast<uint4*>(&Bsh[buf][wv][lane << 4]) = d0;
    *reinterpret_cast<uint4*>(&Bsh[buf][wv][(lane << 4) + 8]) = d1;
#endif
  };

  stage(0, 0);
  for (int kt = 0; kt < KT; ++kt) {
    const int buf = kt & 1;
    const bool more = (kt + 1 < KT);
    if (more) stage(kt + 1, buf ^ 1);

    // A fragments (global, L2-resident state) — overlap with async stage
    const int kofs = kt << 5;
    v16bf af[2];
#pragma unroll
    for (int mi = 0; mi < 2; ++mi) {
      const int b = (mb << 7) + (wv << 5) + (mi << 4) + mrow;
      const __bf16* row = (kofs < K0)
          ? (A0 + (long)b * a0s + kofs)
          : (A1 + (long)b * a1s + (kofs - K0));
      af[mi] = load_a_frag(row, kh);
    }

#if USE_ASYNC_LDS
    if (more) asm volatile("s_wait_asynccnt 0x2" ::: "memory");
    else      asm volatile("s_wait_asynccnt 0x0" ::: "memory");
#endif
    __syncthreads();   // all 4 waves' tiles visible

#pragma unroll
    for (int g = 0; g < 4; ++g) {
      const __bf16* bp = &Bsh[buf][g][lane << 4];
      v16bf bf = load_frag_pair(bp, bp + 8);   // ds_load_b128 x2
#pragma unroll
      for (int mi = 0; mi < 2; ++mi)
        acc[g][mi] = __builtin_amdgcn_wmma_f32_16x16x32_bf16(
            false, af[mi], false, bf, (short)0, acc[g][mi], false, false);
    }
    __syncthreads();   // done reading buf before it is restaged
  }

  // ---- epilogue: i,f,g,o -> conditional (h,c) update ----
  const int nloc  = lane & 15;
  const int mbase = (lane >> 4) << 3;
  const int ng    = (nh << 4) + nloc;
  float bb[4], wc[4];
#pragma unroll
  for (int g = 0; g < 4; ++g) {
    bb[g] = bias[g * HD + ng];
    wc[g] = wxc0 ? wxc0[(long)(g * HD + ng) * wxc0s] : 0.0f;
  }
#pragma unroll
  for (int mi = 0; mi < 2; ++mi) {
#pragma unroll
    for (int r = 0; r < 8; ++r) {
      const int b = (mb << 7) + (wv << 5) + (mi << 4) + mbase + r;
      const float flag = flags[(long)b * flagss];
      float gi = acc[0][mi][r] + bb[0] + flag * wc[0];
      float gf = acc[1][mi][r] + bb[1] + flag * wc[1];
      float gg = acc[2][mi][r] + bb[2] + flag * wc[2];
      float go = acc[3][mi][r] + bb[3] + flag * wc[3];
      gi = sigmoidf_(gi);
      gf = sigmoidf_(gf);
      go = sigmoidf_(go);
      gg = tanhf(gg);
      const long idx = (long)b * HD + ng;
      const float co = c_in[idx], ho = h_in[idx];
      const float cn = gf * co + gi * gg;
      const float hn = go * tanhf(cn);
      const float cw = flag * cn + (1.0f - flag) * co;
      const float hw = flag * hn + (1.0f - flag) * ho;
      c_out[idx] = cw;
      h_out[idx] = hw;
      hbf_out[idx] = (__bf16)hw;
    }
  }
}

// ---- final linear: out = h1 @ Wl.T + bl  (512x128, K=1024) ----------------
__global__ __launch_bounds__(32)
void final_linear_kernel(const __bf16* __restrict__ Abf,   // (512,1024) bf16
                         const __bf16* __restrict__ Wpk,   // packed, NT=8, KT=32
                         const float* __restrict__ bias,   // 128
                         float* __restrict__ out) {        // (512,128)
  const int nt   = blockIdx.x;   // 0..7
  const int mblk = blockIdx.y;   // 0..15
  const int lane = threadIdx.x;
  const int mrow = lane & 15, kh = lane >> 4;

  v8f acc[2];
#pragma unroll
  for (int mi = 0; mi < 2; ++mi) acc[mi] = (v8f)0.0f;

  for (int kt = 0; kt < 32; ++kt) {
    v16bf af[2];
#pragma unroll
    for (int mi = 0; mi < 2; ++mi) {
      const int b = (mblk << 5) + (mi << 4) + mrow;
      af[mi] = load_a_frag(Abf + (long)b * HD + (kt << 5), kh);
    }
    const __bf16* bp = Wpk + (((long)kt * 8 + nt) << 9) + (lane << 4);
    v16bf bf = load_frag_pair(bp, bp + 8);
#pragma unroll
    for (int mi = 0; mi < 2; ++mi)
      acc[mi] = __builtin_amdgcn_wmma_f32_16x16x32_bf16(
          false, af[mi], false, bf, (short)0, acc[mi], false, false);
  }

  const int nloc  = lane & 15;
  const int mbase = (lane >> 4) << 3;
  const int n     = (nt << 4) + nloc;
  const float bl  = bias[n];
#pragma unroll
  for (int mi = 0; mi < 2; ++mi)
#pragma unroll
    for (int r = 0; r < 8; ++r) {
      const int b = (mblk << 5) + (mi << 4) + mbase + r;
      out[(long)b * 128 + n] = acc[mi][r] + bl;
    }
}

// ---------------------------------------------------------------------------
extern "C" void kernel_launch(void* const* d_in, const int* in_sizes, int n_in,
                              void* d_out, int out_size, void* d_ws, size_t ws_size,
                              hipStream_t stream) {
  (void)in_sizes; (void)n_in; (void)out_size; (void)ws_size;
  const float* x   = (const float*)d_in[0];   // (512,64,64)
  const float* Wx0 = (const float*)d_in[1];   // (4096,64)
  const float* Wh0 = (const float*)d_in[2];   // (4096,1024)
  const float* b0  = (const float*)d_in[3];   // (4096,)
  const float* Wx1 = (const float*)d_in[4];   // (4096,1025)
  const float* Wh1 = (const float*)d_in[5];   // (4096,1024)
  const float* b1  = (const float*)d_in[6];   // (4096,)
  const float* Wl  = (const float*)d_in[7];   // (128,1024)
  const float* bl  = (const float*)d_in[8];   // (128,)
  float* out = (float*)d_out;

  // ---- workspace layout ----
  char* base = (char*)d_ws;
  size_t off = 0;
  auto take = [&](size_t bytes) -> char* {
    char* p = base + off;
    off += (bytes + 255) & ~(size_t)255;
    return p;
  };
  const long NX   = (long)BSZ * TSEQ * INF_;      // 2,097,152
  const long SOFT = (long)BSZ * HD;               // 524,288 per state
  __bf16* Xbf  = (__bf16*)take(NX * 2);
  __bf16* W0pk = (__bf16*)take((long)1088 * 4096 * 2);
  __bf16* W1pk = (__bf16*)take((long)2048 * 4096 * 2);
  __bf16* Wlpk = (__bf16*)take((long)1024 * 128 * 2);
  float*  stF[2];  // [h0 | c0 | h1 | c1] each SOFT floats, contiguous
  stF[0] = (float*)take(4 * SOFT * 4);
  stF[1] = (float*)take(4 * SOFT * 4);
  __bf16* stBF[2]; // [h0bf | h1bf] each SOFT bf16
  stBF[0] = (__bf16*)take(2 * SOFT * 2);
  stBF[1] = (__bf16*)take(2 * SOFT * 2);

  auto h0f = [&](int i) { return stF[i]; };
  auto c0f = [&](int i) { return stF[i] + SOFT; };
  auto h1f = [&](int i) { return stF[i] + 2 * SOFT; };
  auto c1f = [&](int i) { return stF[i] + 3 * SOFT; };
  auto h0bf = [&](int i) { return stBF[i]; };
  auto h1bf = [&](int i) { return stBF[i] + SOFT; };

  // ---- prep: conversions + weight packing + state zero-init ----
  cvt_bf16_kernel<<<dim3((NX + 255) / 256), dim3(256), 0, stream>>>(x, Xbf, NX);
  {
    long tot0 = (long)1088 * 4096;   // W0cat = [Wx0.T(64) | Wh0.T(1024)]
    pack_weights_kernel<<<dim3((tot0 + 255) / 256), dim3(256), 0, stream>>>(
        Wx0, 64, 0, 64, Wh0, 1024, W0pk, 256, tot0);
    long tot1 = (long)2048 * 4096;   // W1cat = [Wx1[:,1:].T(1024) | Wh1.T(1024)]
    pack_weights_kernel<<<dim3((tot1 + 255) / 256), dim3(256), 0, stream>>>(
        Wx1, 1025, 1, 1024, Wh1, 1024, W1pk, 256, tot1);
    long totl = (long)1024 * 128;    // Wl.T
    pack_weights_kernel<<<dim3((totl + 255) / 256), dim3(256), 0, stream>>>(
        Wl, 1024, 0, 1024, nullptr, 1024, Wlpk, 8, totl);
  }
  zero_u32_kernel<<<dim3((unsigned)(4 * SOFT / 256)), dim3(256), 0, stream>>>(
      (uint32_t*)stF[0], 4 * SOFT);
  zero_u32_kernel<<<dim3((unsigned)(SOFT / 256)), dim3(256), 0, stream>>>(
      (uint32_t*)stBF[0], SOFT);  // 2*SOFT bf16 == SOFT u32

  const dim3 gateGrid(64, 4), gateBlk(128);
  float* outH0 = out + (long)BSZ * 128;
  float* outC0 = outH0 + SOFT;
  float* outH1 = outC0 + SOFT;
  float* outC1 = outH1 + SOFT;

  // ---- recurrence: 2 fused GEMM+epilogue launches per timestep ----
  for (int t = 0; t < TSEQ; ++t) {
    const int p = t & 1, q = (t + 1) & 1;
    const bool last = (t == TSEQ - 1);
    const float* flg = x + (long)t * INF_;   // flag[b] = x[b][t][0], stride T*IN

    // layer 0: A = [x_t (K=64) | h0 (K=1024)]
    lstm_gate_step<<<gateGrid, gateBlk, 0, stream>>>(
        Xbf + (long)t * INF_, (long)TSEQ * INF_, 64,
        h0bf(p), (long)HD, 1024,
        W0pk, b0, nullptr, 0,
        flg, (long)TSEQ * INF_,
        c0f(p), h0f(p),
        last ? outC0 : c0f(q), last ? outH0 : h0f(q),
        h0bf(q));

    // layer 1: A = [h0_new (K=1024) | h1 (K=1024)], + flag * Wx1[:,0]
    lstm_gate_step<<<gateGrid, gateBlk, 0, stream>>>(
        h0bf(q), (long)HD, 1024,
        h1bf(p), (long)HD, 1024,
        W1pk, b1, Wx1, 1025,
        flg, (long)TSEQ * INF_,
        c1f(p), h1f(p),
        last ? outC1 : c1f(q), last ? outH1 : h1f(q),
        h1bf(q));
  }

  // ---- last_output = h1 @ Wl.T + bl ----
  final_linear_kernel<<<dim3(8, 16), dim3(32), 0, stream>>>(h1bf(0), Wlpk, bl, out);
}
